// LinearAttention_29454885716745
// MI455X (gfx1250) — compile-verified, matrix operand order checked by reading, not run on hardware
//
#include <hip/hip_runtime.h>

// ---------- types ----------
typedef __attribute__((ext_vector_type(16))) __bf16 v16bf;
typedef __attribute__((ext_vector_type(8)))  __bf16 v8bf;
typedef __attribute__((ext_vector_type(8)))  float  v8f;
typedef __attribute__((ext_vector_type(4)))  unsigned int u32x4;
typedef __attribute__((ext_vector_type(8)))  unsigned int u32x8;

#define HEADS 8
#define DHEAD 32
#define NPOS  4096            // 64*64 tokens per batch image
#define BATCH 16
#define CDIM  256
#define HID   256
#define MROWS (BATCH * NPOS)  // 65536
#define LDQKV 768
#define SCALE 0.17677669529663687f  // 32^-0.5
#define EPS   1e-6f

// workspace layout (bytes)
#define WQ_OFF   0u
#define WO_OFF   393216u                  // 768*256*2
#define A_OFF    524288u                  // WO_OFF + 256*256*2
#define QKV_OFF  34078720u                // A_OFF + 65536*256*2
// OutPre reuses A region; Y (f32, 67MB) reuses QKV region (100MB)

// ---------- kernel 0: weights -> bf16 ----------
__global__ __launch_bounds__(256) void cvt_weights(
    const float* __restrict__ wq, const float* __restrict__ wo,
    __bf16* __restrict__ wqb, __bf16* __restrict__ wob) {
  int idx = blockIdx.x * 256 + threadIdx.x;
  if (idx < LDQKV * CDIM) wqb[idx] = (__bf16)wq[idx];
  if (idx < CDIM * HID)   wob[idx] = (__bf16)wo[idx];
}

// ---------- kernel 1: channel-LN of x -> bf16 A[65536, 256] (K contiguous) ----------
__global__ __launch_bounds__(256) void ln_to_bf16(
    const float* __restrict__ x, __bf16* __restrict__ A) {
  __shared__ float ssum[4][64];
  __shared__ float ssq[4][64];
  __shared__ float mean_[64];
  __shared__ float rstd_[64];
  __shared__ __bf16 tile[64][264];  // padded to break bank conflicts

  int t  = threadIdx.x;
  int pl = t & 63;       // local position
  int cg = t >> 6;       // channel group (0..3), 64 channels each
  int p0 = blockIdx.x * 64;
  int b  = p0 >> 12;
  size_t base = ((size_t)b * CDIM) * NPOS + (p0 & (NPOS - 1)) + pl;

  float s = 0.f, sq = 0.f;
  for (int j = 0; j < 64; ++j) {
    float v = x[base + (size_t)(cg * 64 + j) * NPOS];
    s += v; sq += v * v;
  }
  ssum[cg][pl] = s; ssq[cg][pl] = sq;
  __syncthreads();
  if (t < 64) {
    float S = ssum[0][t] + ssum[1][t] + ssum[2][t] + ssum[3][t];
    float Q = ssq[0][t] + ssq[1][t] + ssq[2][t] + ssq[3][t];
    float m = S * (1.f / 256.f);
    mean_[t] = m;
    rstd_[t] = rsqrtf(Q * (1.f / 256.f) - m * m + EPS);
  }
  __syncthreads();
  float m = mean_[pl], r = rstd_[pl];
  for (int j = 0; j < 64; ++j) {
    int c = cg * 64 + j;
    float v = x[base + (size_t)c * NPOS];
    tile[pl][c] = (__bf16)((v - m) * r);
  }
  __syncthreads();
  // flat coalesced write-out: 64 rows x 128 dwords
  int row = t >> 2, seg = t & 3;
  const unsigned int* src = reinterpret_cast<const unsigned int*>(&tile[row][0]);
  unsigned int* dst = reinterpret_cast<unsigned int*>(A + (size_t)(p0 + row) * CDIM);
#pragma unroll
  for (int i = 0; i < 32; ++i) dst[seg * 32 + i] = src[seg * 32 + i];
}

// ---------- WMMA fragment loads (documented 16-bit A/B layout) ----------
__device__ __forceinline__ v16bf frag_ld(const __bf16* __restrict__ base,
                                         int row0, int ld, int k0, int lane) {
  int r  = row0 + (lane & 15);
  int kb = k0 + ((lane >> 4) << 3);           // +8 for upper half-wave
  const __bf16* p = base + (size_t)r * ld + kb;
  v8bf lo = *reinterpret_cast<const v8bf*>(p);       // K = kb .. kb+7
  v8bf hi = *reinterpret_cast<const v8bf*>(p + 16);  // K = kb+16 .. kb+23
  v16bf out;
#pragma unroll
  for (int i = 0; i < 8; ++i) { out[i] = lo[i]; out[8 + i] = hi[i]; }
  return out;
}

__device__ __forceinline__ v16bf frag_ld_lds(const __bf16* base,
                                             int row0, int ld, int k0, int lane) {
  int r  = row0 + (lane & 15);
  int kb = k0 + ((lane >> 4) << 3);
  const __bf16* p = base + r * ld + kb;
  v8bf lo = *reinterpret_cast<const v8bf*>(p);
  v8bf hi = *reinterpret_cast<const v8bf*>(p + 16);
  v16bf out;
#pragma unroll
  for (int i = 0; i < 8; ++i) { out[i] = lo[i]; out[8 + i] = hi[i]; }
  return out;
}

// ---------- kernels 2 & 4: C[M,N] = A[M,256] * B[N,256]^T ----------
// Block = 8 waves = 128 M-rows. A slice register-resident (read once from HBM).
// B panel (64 x 256 bf16 = 32KB) staged per N-group into LDS by the Tensor
// Data Mover (tensor_load_to_lds, TENSORcnt), shared by all 8 waves.
__global__ __launch_bounds__(256) void gemm_wmma_bf16(
    const __bf16* __restrict__ A, const __bf16* __restrict__ B,
    const float* __restrict__ bias,
    float* __restrict__ Cf, __bf16* __restrict__ Cbf,
    int M, int N) {
  constexpr int K = 256;
  __shared__ __attribute__((aligned(16))) __bf16 Bs[64 * K];  // 32 KB

  int lane = threadIdx.x & 31;
  int wave = threadIdx.x >> 5;
  int m0 = (blockIdx.x * 8 + wave) * 16;

  // preload this wave's 16x256 A slice into 64 VGPRs (8 x v16bf)
  v16bf afr[8];
#pragma unroll
  for (int kk = 0; kk < 8; ++kk) afr[kk] = frag_ld(A, m0, K, kk * 32, lane);

  unsigned lds_base = (unsigned)(size_t)(&Bs[0]);  // flat LDS addr: low 32 bits = LDS offset
  int coll = lane & 15;
  int rb   = m0 + ((lane >> 4) << 3);

  for (int n0 = 0; n0 < N; n0 += 64) {
    if (wave == 0) {
      // ---- Tensor DMA descriptor (ISA cdna5 ch.8): 2D tile 256x64, 2B elems ----
      unsigned long long ga = (unsigned long long)(const void*)(B + (size_t)n0 * K);
      u32x4 g0;
      g0[0] = 1u;                                   // count=1, user mode
      g0[1] = lds_base;                             // lds_addr (bytes)
      g0[2] = (unsigned)(ga & 0xFFFFFFFFu);         // global_addr[31:0]
      g0[3] = (unsigned)(ga >> 32) | (2u << 30);    // global_addr[56:32] | type=2
      u32x8 g1;
      g1[0] = 1u << 16;                             // workgroup_mask=0, data_size=1 (2B)
      g1[1] = (unsigned)K << 16;                    // tensor_dim0[15:0] @bit48
      g1[2] = ((unsigned)N << 16) | ((unsigned)K >> 16);   // tensor_dim0 hi | tensor_dim1 lo
      g1[3] = ((unsigned)K << 16) | ((unsigned)N >> 16);   // tile_dim0=256 | tensor_dim1 hi
      g1[4] = 64u;                                  // tile_dim1=64, tile_dim2=0
      g1[5] = (unsigned)K;                          // tensor_dim0_stride lo
      g1[6] = 0u;                                   // stride hi / dim1_stride lo
      g1[7] = 0u;
      asm volatile("tensor_load_to_lds %0, %1" : : "s"(g0), "s"(g1) : "memory");
      __builtin_amdgcn_s_wait_tensorcnt(0);
    }
    __syncthreads();

    v8f acc[4] = {};
#pragma unroll
    for (int kk = 0; kk < 8; ++kk) {
#pragma unroll
      for (int j = 0; j < 4; ++j) {
        v16bf bf = frag_ld_lds(Bs, j * 16, K, kk * 32, lane);
        acc[j] = __builtin_amdgcn_wmma_f32_16x16x32_bf16(
            false, afr[kk], false, bf, (short)0, acc[j], false, false);
      }
    }

#pragma unroll
    for (int j = 0; j < 4; ++j) {
      int col = n0 + j * 16 + coll;
      float bb = bias ? bias[col] : 0.f;
#pragma unroll
      for (int r = 0; r < 8; ++r) {
        float v = acc[j][r] + bb;
        size_t idx = (size_t)(rb + r) * N + col;
        if (Cf) Cf[idx] = v;
        else    Cbf[idx] = (__bf16)v;
      }
    }
    __syncthreads();  // protect Bs before next group's DMA
  }
}

// ---------- kernel 3: linear attention per (b,h) ----------
__global__ __launch_bounds__(256) void attention_kernel(
    const __bf16* __restrict__ QKV, const float* __restrict__ mem_kv,
    __bf16* __restrict__ OutPre) {
  __shared__ float red[8][32];
  __shared__ float m_[32];
  __shared__ float is_[32];
  __shared__ float Kc[32][65];
  __shared__ float Vc[32][65];
  __shared__ float ctx[32][33];

  int t = threadIdx.x;
  int b = blockIdx.x >> 3, h = blockIdx.x & 7;
  size_t bbase = (size_t)b * NPOS;
  int cq = h * DHEAD, ck = 256 + h * DHEAD, cv = 512 + h * DHEAD;
  const float* mk = mem_kv + (size_t)h * DHEAD * 4;
  const float* mv = mem_kv + (size_t)HEADS * DHEAD * 4 + (size_t)h * DHEAD * 4;

  // ---- K softmax over tokens (n = 4 mem + 4096): max then sum-exp ----
  int d = t & 31, sub = t >> 5;
  float mx = -1e30f;
  if (sub == 0)
    for (int j = 0; j < 4; ++j) mx = fmaxf(mx, mk[d * 4 + j]);
  for (int n = sub; n < NPOS; n += 8)
    mx = fmaxf(mx, (float)QKV[(bbase + n) * LDQKV + ck + d]);
  red[sub][d] = mx;
  __syncthreads();
  if (t < 32) {
    float v = red[0][t];
    for (int j = 1; j < 8; ++j) v = fmaxf(v, red[j][t]);
    m_[t] = v;
  }
  __syncthreads();
  float md = m_[d];
  float se = 0.f;
  if (sub == 0)
    for (int j = 0; j < 4; ++j) se += __expf(mk[d * 4 + j] - md);
  for (int n = sub; n < NPOS; n += 8)
    se += __expf((float)QKV[(bbase + n) * LDQKV + ck + d] - md);
  red[sub][d] = se;
  __syncthreads();
  if (t < 32) {
    float v = 0.f;
    for (int j = 0; j < 8; ++j) v += red[j][t];
    is_[t] = 1.f / v;
  }
  __syncthreads();

  // ---- context[d][e] = sum_n softmax_n(K)[d,n] * V[e,n] ----
  int dd = t >> 3, e0 = (t & 7) * 4;
  float acc[4] = {0.f, 0.f, 0.f, 0.f};
  {
    float mdd = m_[dd], iss = is_[dd];
    for (int j = 0; j < 4; ++j) {
      float kd = __expf(mk[dd * 4 + j] - mdd) * iss;
#pragma unroll
      for (int i = 0; i < 4; ++i) acc[i] += kd * mv[(e0 + i) * 4 + j];
    }
  }
  int tok = t & 63, cg2 = t >> 6;
  for (int n0 = 0; n0 < NPOS; n0 += 64) {
    const __bf16* rowp = QKV + (bbase + n0 + tok) * LDQKV;
    v8bf kr = *reinterpret_cast<const v8bf*>(rowp + ck + cg2 * 8);
    v8bf vr = *reinterpret_cast<const v8bf*>(rowp + cv + cg2 * 8);
#pragma unroll
    for (int jj = 0; jj < 8; ++jj) {
      int c = cg2 * 8 + jj;
      Kc[c][tok] = __expf((float)kr[jj] - m_[c]) * is_[c];
      Vc[c][tok] = (float)vr[jj];
    }
    __syncthreads();
#pragma unroll 4
    for (int nn = 0; nn < 64; ++nn) {
      float kd = Kc[dd][nn];
      acc[0] += kd * Vc[e0 + 0][nn];
      acc[1] += kd * Vc[e0 + 1][nn];
      acc[2] += kd * Vc[e0 + 2][nn];
      acc[3] += kd * Vc[e0 + 3][nn];
    }
    __syncthreads();
  }
#pragma unroll
  for (int i = 0; i < 4; ++i) ctx[dd][e0 + i] = acc[i];
  __syncthreads();

  // ---- out[e,n] = sum_d ctx[d,e] * (softmax_d(Q[:,n]) * SCALE) ----
  for (int n = t; n < NPOS; n += 256) {
    const __bf16* rowq = QKV + (bbase + n) * LDQKV + cq;
    float qv[32];
#pragma unroll
    for (int g = 0; g < 4; ++g) {
      v8bf q8 = *reinterpret_cast<const v8bf*>(rowq + g * 8);
#pragma unroll
      for (int jj = 0; jj < 8; ++jj) qv[g * 8 + jj] = (float)q8[jj];
    }
    float qm = qv[0];
#pragma unroll
    for (int i = 1; i < 32; ++i) qm = fmaxf(qm, qv[i]);
    float qs = 0.f;
#pragma unroll
    for (int i = 0; i < 32; ++i) { qv[i] = __expf(qv[i] - qm); qs += qv[i]; }
    float scl = SCALE / qs;
    float o[32];
#pragma unroll
    for (int e = 0; e < 32; ++e) o[e] = 0.f;
#pragma unroll
    for (int i = 0; i < 32; ++i) {
      float qd = qv[i] * scl;
#pragma unroll
      for (int e = 0; e < 32; ++e) o[e] += ctx[i][e] * qd;
    }
    __bf16* op = OutPre + (bbase + n) * HID + cq;
#pragma unroll
    for (int g = 0; g < 4; ++g) {
      v8bf st;
#pragma unroll
      for (int jj = 0; jj < 8; ++jj) st[jj] = (__bf16)o[g * 8 + jj];
      *reinterpret_cast<v8bf*>(op + g * 8) = st;
    }
  }
}

// ---------- kernel 5: channel-LN of Y + residual -> NCHW output ----------
__global__ __launch_bounds__(256) void ln_res_out(
    const float* __restrict__ Y, const float* __restrict__ x,
    float* __restrict__ out) {
  __shared__ float yt[32][257];
  __shared__ float r1[32][8];
  __shared__ float r2[32][8];
  __shared__ float mean_[32];
  __shared__ float rstd_[32];

  int t  = threadIdx.x;
  int p0 = blockIdx.x * 32;
  int b  = p0 >> 12;
  int pix = p0 & (NPOS - 1);

  const float* yrow = Y + (size_t)p0 * CDIM;
  for (int i = 0; i < 32; ++i) yt[i][t] = yrow[(size_t)i * CDIM + t];
  __syncthreads();

  int pos = t >> 3, sb = t & 7;
  float s = 0.f, sq = 0.f;
  for (int j = 0; j < 32; ++j) {
    float v = yt[pos][sb * 32 + j];
    s += v; sq += v * v;
  }
  r1[pos][sb] = s; r2[pos][sb] = sq;
  __syncthreads();
  if (t < 32) {
    float S = 0.f, Q = 0.f;
    for (int j = 0; j < 8; ++j) { S += r1[t][j]; Q += r2[t][j]; }
    float m = S * (1.f / 256.f);
    mean_[t] = m;
    rstd_[t] = rsqrtf(Q * (1.f / 256.f) - m * m + EPS);
  }
  __syncthreads();

  int pl = t & 31, cg = t >> 5;
  float m = mean_[pl], r = rstd_[pl];
  for (int i = 0; i < 32; ++i) {
    int c = i * 8 + cg;
    size_t gi = ((size_t)b * CDIM + c) * NPOS + pix + pl;
    out[gi] = (yt[pl][c] - m) * r + x[gi];
  }
}

// ---------- host launcher ----------
extern "C" void kernel_launch(void* const* d_in, const int* in_sizes, int n_in,
                              void* d_out, int out_size, void* d_ws, size_t ws_size,
                              hipStream_t stream) {
  const float* x      = (const float*)d_in[0];
  const float* mem_kv = (const float*)d_in[1];
  const float* w_qkv  = (const float*)d_in[2];
  const float* w_out  = (const float*)d_in[3];
  const float* b_out  = (const float*)d_in[4];
  float* out = (float*)d_out;

  char* ws = (char*)d_ws;
  __bf16* wq_bf  = (__bf16*)(ws + WQ_OFF);
  __bf16* wo_bf  = (__bf16*)(ws + WO_OFF);
  __bf16* Abf    = (__bf16*)(ws + A_OFF);
  __bf16* QKVbf  = (__bf16*)(ws + QKV_OFF);
  __bf16* OutPre = Abf;                     // reuse: A dead after GEMM1
  float*  Yf     = (float*)(ws + QKV_OFF);  // reuse: qkv dead after attention

  cvt_weights<<<768, 256, 0, stream>>>(w_qkv, w_out, wq_bf, wo_bf);
  ln_to_bf16<<<MROWS / 64, 256, 0, stream>>>(x, Abf);
  gemm_wmma_bf16<<<MROWS / 128, 256, 0, stream>>>(
      Abf, wq_bf, nullptr, nullptr, QKVbf, MROWS, LDQKV);
  attention_kernel<<<BATCH * HEADS, 256, 0, stream>>>(QKVbf, mem_kv, OutPre);
  gemm_wmma_bf16<<<MROWS / 128, 256, 0, stream>>>(
      OutPre, wo_bf, b_out, Yf, nullptr, MROWS, CDIM);
  ln_res_out<<<MROWS / 32, 256, 0, stream>>>(Yf, x, out);
}
